// Relative_AttentionHead_7799660610318
// MI455X (gfx1250) — matrix operand order repaired and checked by reading, not compiled, not run on hardware
//
#include <hip/hip_runtime.h>
#include <hip/hip_bf16.h>

typedef __attribute__((ext_vector_type(16))) _Float16 v16h;
typedef __attribute__((ext_vector_type(8)))  _Float16 v8h;
typedef __attribute__((ext_vector_type(8)))  float    v8f;
typedef __attribute__((ext_vector_type(4)))  unsigned v4u;
typedef __attribute__((ext_vector_type(8)))  int      v8i;
typedef __attribute__((ext_vector_type(4)))  int      v4i;

#define NSEQ 2048
#define NBATCH 8
#define DMODEL 512
#define KD 64
#define NF 70
#define NEG_BIG (-3.0e38f)

#if __has_builtin(__builtin_amdgcn_tensor_load_to_lds)
#define USE_TDM 1
#else
#define USE_TDM 0
#endif

static __device__ __forceinline__ v8f wmma_f16(v16h a, v16h b, v8f c) {
  return __builtin_amdgcn_wmma_f32_16x16x32_f16(
      false, a, false, b, (short)0, c, false, false);
}

static __device__ __forceinline__ v8h cvt8(float4 a, float4 b) {
  v8h h;
  h[0] = (_Float16)a.x; h[1] = (_Float16)a.y; h[2] = (_Float16)a.z; h[3] = (_Float16)a.w;
  h[4] = (_Float16)b.x; h[5] = (_Float16)b.y; h[6] = (_Float16)b.z; h[7] = (_Float16)b.w;
  return h;
}

// ---- gfx1250 async global->LDS copy (ASYNCcnt-tracked) -----------------------
static __device__ __forceinline__ void async_ld_b128(unsigned lds_byte_addr,
                                                     const void* gptr) {
  asm volatile("global_load_async_to_lds_b128 %0, %1, off"
               :: "v"(lds_byte_addr), "v"((unsigned long long)(size_t)gptr)
               : "memory");
}
static __device__ __forceinline__ void wait_async0() {
#if __has_builtin(__builtin_amdgcn_s_wait_asynccnt)
  __builtin_amdgcn_s_wait_asynccnt(0);
#else
  asm volatile("s_wait_asynccnt 0x0" ::: "memory");
#endif
}

#if USE_TDM
// ---- gfx1250 Tensor Data Mover: 2-D tile Global -> LDS -----------------------
// D# per CDNA5 ISA. data_size = 2 bytes (f16).
// dim0 = contiguous axis; tile is tile_d0 x tile_d1 at gaddr (tile start).
// Toolchain = 6-arg builtin: (v4u g0, v8i g1, v4i, v4i, v8i, i32 cpol).
static __device__ __forceinline__ void tdm_load_2d(
    unsigned lds_addr, const void* gaddr,
    unsigned tensor_d0, unsigned tensor_d1,
    unsigned tile_d0, unsigned tile_d1, unsigned stride0)
{
  const unsigned long long ga = (unsigned long long)(size_t)gaddr;
  v4u g0;
  g0[0] = 1u;                                            // count=1, user D#
  g0[1] = lds_addr;                                      // lds_addr [63:32]
  g0[2] = (unsigned)(ga & 0xffffffffu);                  // global_addr[31:0]
  g0[3] = (unsigned)((ga >> 32) & 0x01ffffffu)           // global_addr[56:32]
        | (2u << 30);                                    // type = 2 ("image")
  v8i g1;
  g1[0] = 0x00010000;                                    // data_size=1 (2B)
  g1[1] = (int)((tensor_d0 & 0xffffu) << 16);            // tensor_dim0[15:0]
  g1[2] = (int)((tensor_d0 >> 16) | ((tensor_d1 & 0xffffu) << 16));
  g1[3] = (int)((tensor_d1 >> 16) | (tile_d0 << 16));    // tile_dim0
  g1[4] = (int)(tile_d1 & 0xffffu);                      // tile_dim1; tile_dim2=0
  g1[5] = (int)stride0;                                  // tensor_dim0_stride
  g1[6] = 0;
  g1[7] = 0;
  const v4i z4 = (v4i)0;                                 // groups 2/3 unused (2D)
  const v8i z8 = (v8i)0;
  __builtin_amdgcn_tensor_load_to_lds(g0, g1, z4, z4, z8, 0);
}
static __device__ __forceinline__ void wait_tensor0() {
  __builtin_amdgcn_s_wait_tensorcnt(0);
}
#endif

// ---------------------------------------------------------------------------
// Kernel 1: batch-invariant trig bias B[n,m] (computed once). Trans-op heavy.
// ---------------------------------------------------------------------------
__global__ __launch_bounds__(256) void bias_kernel(
    const float* __restrict__ bt,    // [N,N,4]
    const float* __restrict__ Wr,    // [F,2]
    const float* __restrict__ Cc,    // [F]
    const float* __restrict__ logp,  // [1]
    float* __restrict__ biasB)       // [N,N]
{
  __shared__ float sW[2 * NF];
  __shared__ float sC[NF];
  int tid = threadIdx.x;
  if (tid < 2 * NF) sW[tid] = Wr[tid];
  if (tid < NF)     sC[tid] = Cc[tid];
  __syncthreads();

  size_t idx = (size_t)blockIdx.x * 256 + tid;
  const float4 bv = *(const float4*)(bt + idx * 4);
  float p = fminf(fmaxf(__expf(logp[0]), 0.5f), 3.0f);

  float sum = 0.0f;
  #pragma unroll 2
  for (int f = 0; f < NF; ++f) {
    float w0 = sW[2 * f], w1 = sW[2 * f + 1];
    float k1 = fmaf(bv.x, w0, bv.y * w1);
    float k2 = fmaf(bv.z, w0, bv.w * w1);
    float d = (__cosf(k1) + __sinf(k1)) - (__cosf(k2) + __sinf(k2));
    sum = fmaf(d, sC[f], sum);
  }
  float sigma = fabsf(sum) + 1e-6f;
  biasB[idx] = __expf(-__powf(sigma, p));
}

// ---------------------------------------------------------------------------
// Kernel 2: Q/K/V projections with WMMA f16 (W chunk staged in LDS per block).
// ---------------------------------------------------------------------------
__global__ __launch_bounds__(256) void qkv_kernel(
    const float* __restrict__ src,   // [B*N][512]
    const float* __restrict__ Wq,    // [64][512]
    const float* __restrict__ Wk,
    const float* __restrict__ Wv,
    _Float16* __restrict__ Qh,       // [B*N][64]
    _Float16* __restrict__ Kh,       // [B*N][64]
    _Float16* __restrict__ Vth)      // [B][64][2048]
{
  __shared__ __align__(32) _Float16 sW[3 * 64 * 32];

  const int tid  = threadIdx.x;
  const int lane = tid & 31;
  const int wave = tid >> 5;
  const int tile = blockIdx.x * 8 + wave;
  const int row0 = tile * 16;
  const int l16  = lane & 15;
  const int half = lane >> 4;

  const int sk  = tid >> 2;
  const int sd8 = (tid & 3) * 8;

  v8f acc[12];
  #pragma unroll
  for (int i = 0; i < 12; ++i) acc[i] = (v8f)0.f;

  for (int dc = 0; dc < DMODEL; dc += 32) {
    #pragma unroll
    for (int mi = 0; mi < 3; ++mi) {
      const float* wm = (mi == 0) ? Wq : (mi == 1) ? Wk : Wv;
      const float* wp = wm + (size_t)sk * DMODEL + dc + sd8;
      float4 f0 = *(const float4*)(wp);
      float4 f1 = *(const float4*)(wp + 4);
      *(v8h*)(sW + mi * 2048 + sk * 32 + sd8) = cvt8(f0, f1);
    }
    __syncthreads();

    const float* ap = src + (size_t)(row0 + l16) * DMODEL + dc + half * 16;
    float4 a0 = *(const float4*)(ap);
    float4 a1 = *(const float4*)(ap + 4);
    float4 a2 = *(const float4*)(ap + 8);
    float4 a3 = *(const float4*)(ap + 12);
    v8h alo = cvt8(a0, a1), ahi = cvt8(a2, a3);
    v16h a = __builtin_shufflevector(alo, ahi, 0,1,2,3,4,5,6,7,8,9,10,11,12,13,14,15);

    #pragma unroll
    for (int mi = 0; mi < 3; ++mi) {
      #pragma unroll
      for (int kg = 0; kg < 4; ++kg) {
        const _Float16* wl = sW + mi * 2048 + (kg * 16 + l16) * 32 + half * 16;
        acc[mi * 4 + kg] = wmma_f16(a, *(const v16h*)(wl), acc[mi * 4 + kg]);
      }
    }
    __syncthreads();
  }

  const int bidx = row0 >> 11;
  const int mloc = row0 & (NSEQ - 1);
  #pragma unroll
  for (int kg = 0; kg < 4; ++kg) {
    const int k = kg * 16 + l16;
    #pragma unroll
    for (int r = 0; r < 8; ++r) {
      const int grow = row0 + half * 8 + r;
      const size_t qa = (size_t)grow * KD + k;
      Qh[qa] = (_Float16)acc[0 * 4 + kg][r];
      Kh[qa] = (_Float16)acc[1 * 4 + kg][r];
      Vth[((size_t)bidx * KD + k) * NSEQ + (mloc + half * 8 + r)] =
          (_Float16)acc[2 * 4 + kg][r];
    }
  }
}

// ---------------------------------------------------------------------------
// Kernel 3: flash attention; K/V tiles staged by TDM (or per-thread async),
// double-buffered; S^T = K*Q^T so softmax lands in WMMA A-layout for P*V.
// ---------------------------------------------------------------------------
__global__ __launch_bounds__(128) void flash_kernel(
    const float*    __restrict__ biasB,  // [N][N]
    const float*    __restrict__ mult,   // [B][N][N]
    const _Float16* __restrict__ Qh,     // [B*N][64]
    const _Float16* __restrict__ Kh,     // [B*N][64]
    const _Float16* __restrict__ Vth,    // [B][64][N]
    const float*    __restrict__ sPtr,   // [1]
    float*          __restrict__ out)    // [B][N][64]
{
  __shared__ __align__(32) _Float16 sK[2 * 32 * 64];
  __shared__ __align__(32) _Float16 sV[2 * 64 * 32];

  const int b    = blockIdx.y;
  const int tid  = threadIdx.x;
  const int lane = tid & 31;
  const int wave = tid >> 5;
  const int l16  = lane & 15;
  const int half = lane >> 4;
  const int n0   = blockIdx.x * 64 + wave * 16;

  const float sscale = 0.125f * sPtr[0];

  const _Float16* qrow = Qh + ((size_t)b * NSEQ + n0 + l16) * KD;
  const v16h qb0 = *(const v16h*)(qrow + half * 16);
  const v16h qb1 = *(const v16h*)(qrow + 32 + half * 16);

  v8f o0 = (v8f)0.f, o1 = (v8f)0.f, o2 = (v8f)0.f, o3 = (v8f)0.f;
  float mrun = NEG_BIG, lrun = 0.0f;

  const float* bp = biasB + (size_t)(n0 + l16) * NSEQ;
  const float* mp = mult + ((size_t)b * NSEQ + n0 + l16) * NSEQ;

  const _Float16* gKbase = Kh + (size_t)b * NSEQ * KD;   // [2048][64]
  const _Float16* gVbase = Vth + (size_t)b * KD * NSEQ;  // [64][2048]

#if USE_TDM
  // wave 0 drives the Tensor Data Mover (TDM ignores EXEC -> one issuer only)
  if (wave == 0) {
    tdm_load_2d((unsigned)(size_t)sK, gKbase, KD, NSEQ, KD, 32, KD);
    tdm_load_2d((unsigned)(size_t)sV, gVbase, NSEQ, KD, 32, KD, NSEQ);
  }
#else
  const int kmr = tid >> 2, kcb = (tid & 3) * 16;
  const int vkr = tid >> 1, vcb = (tid & 1) * 16;
  const _Float16* gK = gKbase + (size_t)kmr * KD + kcb;
  const _Float16* gV = gVbase + (size_t)vkr * NSEQ + vcb;
  const unsigned lK0 = (unsigned)(size_t)(sK + kmr * 64 + kcb);
  const unsigned lV0 = (unsigned)(size_t)(sV + vkr * 32 + vcb);
  async_ld_b128(lK0, gK);
  async_ld_b128(lV0, gV);
#endif

  for (int m0 = 0; m0 < NSEQ; m0 += 32) {
    const int cur = (m0 >> 5) & 1;
#if USE_TDM
    if (wave == 0) wait_tensor0();
#else
    wait_async0();
#endif
    __syncthreads();   // staged data visible to all; prev compute done w/ other buf

    if (m0 + 32 < NSEQ) {
      const int nxt = cur ^ 1;
#if USE_TDM
      if (wave == 0) {
        tdm_load_2d((unsigned)(size_t)(sK + nxt * 32 * 64),
                    gKbase + (size_t)(m0 + 32) * KD, KD, NSEQ, KD, 32, KD);
        tdm_load_2d((unsigned)(size_t)(sV + nxt * 64 * 32),
                    gVbase + (m0 + 32), NSEQ, KD, 32, KD, NSEQ);
      }
#else
      async_ld_b128(lK0 + (unsigned)(nxt * 32 * 64 * 2), gK + (size_t)(m0 + 32) * KD);
      async_ld_b128(lV0 + (unsigned)(nxt * 64 * 32 * 2), gV + (m0 + 32));
#endif
      __builtin_prefetch(bp + m0 + 32 + half * 8, 0, 0);
      __builtin_prefetch(mp + m0 + 32 + half * 8, 0, 0);
    }

    const _Float16* bK = sK + cur * (32 * 64);
    const _Float16* bV = sV + cur * (64 * 32);

    const _Float16* kp0 = bK + l16 * 64 + half * 16;
    const _Float16* kp1 = bK + (16 + l16) * 64 + half * 16;
    v8h t0 = *(const v8h*)(kp0),      t1 = *(const v8h*)(kp0 + 8);
    v8h t2 = *(const v8h*)(kp0 + 32), t3 = *(const v8h*)(kp0 + 40);
    v16h a0c0 = __builtin_shufflevector(t0, t1, 0,1,2,3,4,5,6,7,8,9,10,11,12,13,14,15);
    v16h a0c1 = __builtin_shufflevector(t2, t3, 0,1,2,3,4,5,6,7,8,9,10,11,12,13,14,15);
    t0 = *(const v8h*)(kp1);      t1 = *(const v8h*)(kp1 + 8);
    t2 = *(const v8h*)(kp1 + 32); t3 = *(const v8h*)(kp1 + 40);
    v16h a1c0 = __builtin_shufflevector(t0, t1, 0,1,2,3,4,5,6,7,8,9,10,11,12,13,14,15);
    v16h a1c1 = __builtin_shufflevector(t2, t3, 0,1,2,3,4,5,6,7,8,9,10,11,12,13,14,15);

    v8f s0 = (v8f)0.f, s1 = (v8f)0.f;
    s0 = wmma_f16(a0c0, qb0, s0);
    s0 = wmma_f16(a0c1, qb1, s0);
    s1 = wmma_f16(a1c0, qb0, s1);
    s1 = wmma_f16(a1c1, qb1, s1);

    const int mA = m0 + half * 8;
    const int mB = m0 + 16 + half * 8;
    float lg0[8], lg1[8];
    #pragma unroll
    for (int r = 0; r < 8; ++r) {
      lg0[r] = (s0[r] + bp[mA + r]) * sscale * mp[mA + r];
      lg1[r] = (s1[r] + bp[mB + r]) * sscale * mp[mB + r];
    }

    float tmax = lg0[0];
    #pragma unroll
    for (int r = 1; r < 8; ++r) tmax = fmaxf(tmax, lg0[r]);
    #pragma unroll
    for (int r = 0; r < 8; ++r) tmax = fmaxf(tmax, lg1[r]);
    tmax = fmaxf(tmax, __shfl_xor(tmax, 16, 32));
    const float mnew  = fmaxf(mrun, tmax);
    const float scale = __expf(mrun - mnew);

    float ls = 0.0f;
    v16h pa;
    #pragma unroll
    for (int r = 0; r < 8; ++r) {
      const float e0 = __expf(lg0[r] - mnew);
      const float e1 = __expf(lg1[r] - mnew);
      ls += e0 + e1;
      pa[r]     = (_Float16)e0;
      pa[8 + r] = (_Float16)e1;
    }
    ls += __shfl_xor(ls, 16, 32);
    lrun = lrun * scale + ls;
    mrun = mnew;

    #pragma unroll
    for (int r = 0; r < 8; ++r) {
      const float sc = __shfl(scale, half * 8 + r, 32);
      o0[r] *= sc; o1[r] *= sc; o2[r] *= sc; o3[r] *= sc;
    }

    const int vbase = l16 * 32 + half * 8;
    o0 = wmma_f16(pa, __builtin_shufflevector(*(const v8h*)(bV + vbase), *(const v8h*)(bV + vbase + 16), 0,1,2,3,4,5,6,7,8,9,10,11,12,13,14,15), o0);
    o1 = wmma_f16(pa, __builtin_shufflevector(*(const v8h*)(bV + 16 * 32 + vbase), *(const v8h*)(bV + 16 * 32 + vbase + 16), 0,1,2,3,4,5,6,7,8,9,10,11,12,13,14,15), o1);
    o2 = wmma_f16(pa, __builtin_shufflevector(*(const v8h*)(bV + 32 * 32 + vbase), *(const v8h*)(bV + 32 * 32 + vbase + 16), 0,1,2,3,4,5,6,7,8,9,10,11,12,13,14,15), o2);
    o3 = wmma_f16(pa, __builtin_shufflevector(*(const v8h*)(bV + 48 * 32 + vbase), *(const v8h*)(bV + 48 * 32 + vbase + 16), 0,1,2,3,4,5,6,7,8,9,10,11,12,13,14,15), o3);
  }

  const float il = 1.0f / lrun;
  #pragma unroll
  for (int r = 0; r < 8; ++r) {
    const float ilr = __shfl(il, half * 8 + r, 32);
    const int nrow = n0 + half * 8 + r;
    float* op = out + ((size_t)b * NSEQ + nrow) * KD;
    op[l16]      = o0[r] * ilr;
    op[16 + l16] = o1[r] * ilr;
    op[32 + l16] = o2[r] * ilr;
    op[48 + l16] = o3[r] * ilr;
  }
}

// ---------------------------------------------------------------------------
extern "C" void kernel_launch(void* const* d_in, const int* in_sizes, int n_in,
                              void* d_out, int out_size, void* d_ws, size_t ws_size,
                              hipStream_t stream) {
  const float* src  = (const float*)d_in[0];
  const float* bt   = (const float*)d_in[1];
  const float* mult = (const float*)d_in[2];
  const float* Wr   = (const float*)d_in[3];
  const float* Cc   = (const float*)d_in[4];
  const float* logp = (const float*)d_in[5];
  const float* Wq   = (const float*)d_in[6];
  const float* Wk   = (const float*)d_in[7];
  const float* Wv   = (const float*)d_in[8];
  const float* s    = (const float*)d_in[9];
  float* out = (float*)d_out;

  char* ws = (char*)d_ws;
  float*    biasB = (float*)ws;                                   // 16 MB
  _Float16* Qh    = (_Float16*)(ws + (size_t)NSEQ * NSEQ * 4);
  _Float16* Kh    = Qh + (size_t)NBATCH * NSEQ * KD;
  _Float16* Vth   = Kh + (size_t)NBATCH * NSEQ * KD;

  bias_kernel<<<(NSEQ * NSEQ) / 256, 256, 0, stream>>>(bt, Wr, Cc, logp, biasB);
  qkv_kernel<<<(NBATCH * NSEQ / 16) / 8, 256, 0, stream>>>(src, Wq, Wk, Wv, Qh, Kh, Vth);
  flash_kernel<<<dim3(NSEQ / 64, NBATCH), 128, 0, stream>>>(biasB, mult, Qh, Kh, Vth, s, out);
}
